// Mix9sNet_52553219834299
// MI455X (gfx1250) — matrix operand order repaired and checked by reading, not compiled, not run on hardware
//
#include <hip/hip_runtime.h>

typedef __bf16 bf16;
typedef __attribute__((ext_vector_type(16))) __bf16 v16bf;
typedef __attribute__((ext_vector_type(8)))  __bf16 v8bf;
typedef __attribute__((ext_vector_type(8)))  float  v8f;

#define NB    256          // batch
#define CMID  128
#define COUT  64
#define NP    225          // 15*15 pixels
#define NPT   15           // pixel tiles of 16 (padded to 240)
#define PADW  17           // cols -1..15
#define PADH  18           // rows -1..16
#define PPIX  (PADH*PADW)  // 306 padded pixels

// padded-pixel address deltas for taps (w0, w1, w2) per direction:
// delta = di*PADW + dj ; dir0 H, dir1 V, dir2 anti-diag, dir3 main diag
__constant__ int SHIFT_TAB[4][3] = {
  { -1,        0,  1        },
  { -PADW,     0,  PADW     },
  {  PADW - 1, 0, -(PADW-1) },
  { -(PADW+1), 0,  PADW + 1 }
};
__constant__ int DI0[4] = { 0, -1,  1, -1};
__constant__ int DJ0[4] = {-1,  0, -1, -1};
__constant__ int DI2[4] = { 0,  1, -1,  1};
__constant__ int DJ2[4] = { 1,  0,  1,  1};

__device__ __forceinline__ float silu_f(float x) {
  return x / (1.0f + __expf(-x));
}

// GEMM: dst[o,p] = act( sum_k W[o,k] * src_k[p] + bias[o] ) (+ dst[o,p] if RES)
// DIR=true: K = 3*CMID with per-tap spatial shift of the B operand.
template<bool DIR, bool RES>
__device__ __forceinline__ void gemm_lds(
    const bf16* __restrict__ src, bf16* __restrict__ dst,
    const float* __restrict__ w, const float* __restrict__ bias,
    const int* shifts, int lane, int wave)
{
  const int laneLo = lane & 15;
  const int laneHi = lane >> 4;
  const int oBase  = wave * 16;
  const int o0     = oBase + laneHi * 8;   // C/D rows: M = r + 8*laneHi

  v8f acc[NPT];
  #pragma unroll
  for (int pt = 0; pt < NPT; ++pt)
    #pragma unroll
    for (int r = 0; r < 8; ++r)
      acc[pt][r] = bias[o0 + r];

  const int K = DIR ? 3 * CMID : CMID;
  const int kIters = K / 32;

  for (int kt = 0; kt < kIters; ++kt) {
    // ---- A fragment (weights, 16x32 bf16): per lane M=laneLo,
    // K(e) = kt*32 + laneHi*16 + e
    v16bf afrag;
    const int kbA = kt * 32 + laneHi * 16;
    const float* wr = w + (oBase + laneLo) * K;
    #pragma unroll
    for (int e = 0; e < 8; ++e) {
      const int k0 = kbA + e;
      const int k1 = kbA + 8 + e;
      float f0, f1;
      if (DIR) {
        const int tap = kt >> 2;                  // both runs stay in one tap block
        f0 = wr[(k0 - tap * CMID) * 3 + tap];     // layout [Cout][Cin][3]
        f1 = wr[(k1 - tap * CMID) * 3 + tap];
      } else {
        f0 = wr[k0];
        f1 = wr[k1];
      }
      afrag[e]     = (bf16)f0;
      afrag[e + 8] = (bf16)f1;
    }

    // ---- B fragment source: K(e) = kt*32 + laneHi*16 + e, N = laneLo
    const int kRow = kt * 32 + laneHi * 16;
    int cB = kRow, dPix = 0;
    if (DIR) {
      const int tapB = kRow >> 7;                 // 16-run stays in one tap block
      cB   = kRow - tapB * CMID;
      dPix = shifts[tapB];
    }
    #pragma unroll
    for (int pt = 0; pt < NPT; ++pt) {
      const int p = pt * 16 + laneLo;
      const int i = p / 15, j = p - i * 15;       // i may be 15 for pad lanes
      const int pix = (i + 1) * PADW + (j + 1) + dPix;
      const v8bf* bp = (const v8bf*)(src + pix * CMID + cB);
      const v8bf blo = bp[0];
      const v8bf bhi = bp[1];
      const v16bf bfrag = __builtin_shufflevector(blo, bhi,
          0, 1, 2, 3, 4, 5, 6, 7, 8, 9, 10, 11, 12, 13, 14, 15);
      acc[pt] = __builtin_amdgcn_wmma_f32_16x16x32_bf16(
          false, afrag, false, bfrag, (short)0, acc[pt], false, false);
    }
  }

  // ---- epilogue: silu (+in-place residual) -> dst (bf16) ----
  #pragma unroll
  for (int pt = 0; pt < NPT; ++pt) {
    const int p = pt * 16 + laneLo;
    if (p < NP) {
      const int i = p / 15, j = p - i * 15;
      bf16* dp = dst + ((i + 1) * PADW + (j + 1)) * CMID + o0;
      #pragma unroll
      for (int r = 0; r < 8; ++r) {
        float v = silu_f(acc[pt][r]);
        if (RES) v += (float)dp[r];
        dp[r] = (bf16)v;
      }
    }
  }
}

// final 1x1: 128 -> 64, fp32 out, no activation; waves 0..3 active
__device__ __forceinline__ void gemm_out(
    const bf16* __restrict__ src, float* __restrict__ outp,
    const float* __restrict__ w, const float* __restrict__ bias,
    int lane, int wave)
{
  if (wave >= 4) return;
  const int laneLo = lane & 15;
  const int laneHi = lane >> 4;
  const int oBase  = wave * 16;
  const int o0     = oBase + laneHi * 8;

  v8f acc[NPT];
  #pragma unroll
  for (int pt = 0; pt < NPT; ++pt)
    #pragma unroll
    for (int r = 0; r < 8; ++r)
      acc[pt][r] = bias[o0 + r];

  for (int kt = 0; kt < 4; ++kt) {
    v16bf afrag;
    const int kbA = kt * 32 + laneHi * 16;
    const float* wr = w + (oBase + laneLo) * CMID;
    #pragma unroll
    for (int e = 0; e < 8; ++e) {
      afrag[e]     = (bf16)wr[kbA + e];
      afrag[e + 8] = (bf16)wr[kbA + 8 + e];
    }
    const int cB = kt * 32 + laneHi * 16;
    #pragma unroll
    for (int pt = 0; pt < NPT; ++pt) {
      const int p = pt * 16 + laneLo;
      const int i = p / 15, j = p - i * 15;
      const int pix = (i + 1) * PADW + (j + 1);
      const v8bf* bp = (const v8bf*)(src + pix * CMID + cB);
      const v8bf blo = bp[0];
      const v8bf bhi = bp[1];
      const v16bf bfrag = __builtin_shufflevector(blo, bhi,
          0, 1, 2, 3, 4, 5, 6, 7, 8, 9, 10, 11, 12, 13, 14, 15);
      acc[pt] = __builtin_amdgcn_wmma_f32_16x16x32_bf16(
          false, afrag, false, bfrag, (short)0, acc[pt], false, false);
    }
  }

  #pragma unroll
  for (int pt = 0; pt < NPT; ++pt) {
    const int p = pt * 16 + laneLo;
    if (p < NP) {
      #pragma unroll
      for (int r = 0; r < 8; ++r)
        outp[(o0 + r) * NP + p] = acc[pt][r];
    }
  }
}

__global__ __launch_bounds__(256)
void mix9s_kernel(
    const float* __restrict__ x,
    const float* __restrict__ w0,   const float* __restrict__ b0,
    const float* __restrict__ rw,   const float* __restrict__ rb,
    const float* __restrict__ pw,   const float* __restrict__ pb,
    const float* __restrict__ c0w1, const float* __restrict__ c0b1,
    const float* __restrict__ c0w2, const float* __restrict__ c0b2,
    const float* __restrict__ fw,   const float* __restrict__ fb,
    float* __restrict__ out)
{
  __shared__ __align__(16) bf16 bufA[PPIX * CMID];  // 76.5 KB
  __shared__ __align__(16) bf16 bufB[PPIX * CMID];  // 76.5 KB

  const int tid  = threadIdx.x;
  const int lane = tid & 31;
  const int wave = tid >> 5;
  const int b    = blockIdx.x >> 2;
  const int d    = blockIdx.x & 3;

  // zero both buffers once: padding rows/cols must stay zero
  {
    unsigned int* za = (unsigned int*)bufA;
    unsigned int* zb = (unsigned int*)bufB;
    const int ndw = PPIX * CMID / 2;
    for (int t = tid; t < ndw; t += 256) { za[t] = 0u; zb[t] = 0u; }
  }
  __syncthreads();

  // ---- layer 0: directional sparse conv Cin=2 -> 128 (tiny, scalar) ----
  {
    const float* xp = x + (size_t)b * 2 * NP;
    const int di0 = DI0[d], dj0 = DJ0[d], di2 = DI2[d], dj2 = DJ2[d];
    for (int idx = tid; idx < CMID * NP; idx += 256) {
      const int o = idx / NP;
      const int p = idx - o * NP;
      const int i = p / 15, j = p - i * 15;
      float a = b0[o];
      #pragma unroll
      for (int c = 0; c < 2; ++c) {
        const float* wr = w0 + (o * 2 + c) * 3;
        a += wr[1] * xp[c * NP + p];
        const int i0 = i + di0, j0 = j + dj0;
        if (i0 >= 0 && i0 < 15 && j0 >= 0 && j0 < 15)
          a += wr[0] * xp[c * NP + i0 * 15 + j0];
        const int i2 = i + di2, j2 = j + dj2;
        if (i2 >= 0 && i2 < 15 && j2 >= 0 && j2 < 15)
          a += wr[2] * xp[c * NP + i2 * 15 + j2];
      }
      a = silu_f(a);
      bufA[((i + 1) * PADW + (j + 1)) * CMID + o] = (bf16)a;
    }
  }
  __syncthreads();

  const int* shifts = SHIFT_TAB[d];

  // ---- 4 directional residual blocks ----
  for (int l = 0; l < 4; ++l) {
    gemm_lds<true,  false>(bufA, bufB, rw + (size_t)l * CMID * CMID * 3,
                           rb + l * CMID, shifts, lane, wave);
    __syncthreads();
    gemm_lds<false, true >(bufB, bufA, pw + (size_t)l * CMID * CMID,
                           pb + l * CMID, nullptr, lane, wave);
    __syncthreads();
  }

  // ---- Conv0d residual block (two 1x1) ----
  gemm_lds<false, false>(bufA, bufB, c0w1, c0b1, nullptr, lane, wave);
  __syncthreads();
  gemm_lds<false, true >(bufB, bufA, c0w2, c0b2, nullptr, lane, wave);
  __syncthreads();

  // ---- final 1x1 -> out[b][d][64][15][15] fp32 ----
  gemm_out(bufA, out + (size_t)(b * 4 + d) * COUT * NP, fw, fb, lane, wave);
}

extern "C" void kernel_launch(void* const* d_in, const int* in_sizes, int n_in,
                              void* d_out, int out_size, void* d_ws, size_t ws_size,
                              hipStream_t stream) {
  (void)in_sizes; (void)n_in; (void)out_size; (void)d_ws; (void)ws_size;
  const float* x    = (const float*)d_in[0];
  const float* w0   = (const float*)d_in[1];
  const float* b0   = (const float*)d_in[2];
  const float* rw   = (const float*)d_in[3];
  const float* rb   = (const float*)d_in[4];
  const float* pwv  = (const float*)d_in[5];
  const float* pb   = (const float*)d_in[6];
  const float* c0w1 = (const float*)d_in[7];
  const float* c0b1 = (const float*)d_in[8];
  const float* c0w2 = (const float*)d_in[9];
  const float* c0b2 = (const float*)d_in[10];
  const float* fw   = (const float*)d_in[11];
  const float* fb   = (const float*)d_in[12];
  float* out = (float*)d_out;

  dim3 grid(NB * 4);
  dim3 block(256);
  hipLaunchKernelGGL(mix9s_kernel, grid, block, 0, stream,
                     x, w0, b0, rw, rb, pwv, pb,
                     c0w1, c0b1, c0w2, c0b2, fw, fb, out);
}
